// DynamicGatedCrossAttention_77189152244304
// MI455X (gfx1250) — matrix operand order repaired and checked by reading, not compiled, not run on hardware
//
#include <hip/hip_runtime.h>

typedef __attribute__((ext_vector_type(16))) __bf16 v16bf;
typedef __attribute__((ext_vector_type(8)))  float  v8f;
typedef __attribute__((ext_vector_type(4)))  unsigned u32x4;
typedef __attribute__((ext_vector_type(8)))  unsigned u32x8;

#define NPIX 4096
#define CCH  256
#define DQK  32
#define NB   8

// K index held by (half-vector-elem h, lane-half g) for a bf16 16x32 A fragment
// (contiguous: lanes0-15 K=0..15 / lanes16-31 K=16..31, matching the B fills)
__device__ __forceinline__ int a_kidx(int h, int g) {
  return (g << 4) + h;
}

// LDS byte offset of a __shared__ pointer (generic -> addrspace(3) -> int)
#define LDS_OFF(p) ((unsigned)(unsigned long long)(__attribute__((address_space(3))) const __bf16*)(p))

// Issue one TDM descriptor (2 SGPR groups: tensors up to 2D; VADDR2/3 = NULL).
// Group0: [1:0]count=1, [63:32]lds_addr, [120:64]global_addr, [127:126]type=2.
__device__ __forceinline__ void tdm_issue(unsigned lds, unsigned long long gaddr,
                                          u32x8 g1) {
  u32x4 g0;
  g0[0] = 1u;
  g0[1] = lds;
  g0[2] = (unsigned)gaddr;
  g0[3] = ((unsigned)(gaddr >> 32) & 0x01FFFFFFu) | 0x80000000u;
  asm volatile("tensor_load_to_lds %0, %1" :: "s"(g0), "s"(g1) : "memory");
}

// ---------------- projection: qT,kT = (X^T Wq^T), pixel-major [N,32] ----------------
__global__ void __launch_bounds__(32) proj_qk_kernel(
    const float* __restrict__ opt, const float* __restrict__ dem,
    const float* __restrict__ Wq, const float* __restrict__ bq,
    const float* __restrict__ Wk, const float* __restrict__ bk,
    __bf16* __restrict__ qT, __bf16* __restrict__ kT)
{
  const int lane = threadIdx.x & 31;
  const int m15  = lane & 15;
  const int g    = lane >> 4;
  const int b    = blockIdx.y;
  const int i0   = blockIdx.x * 16;
  const float* xo = opt + (size_t)b * CCH * NPIX;
  const float* xd = dem + (size_t)b * CCH * NPIX;

  v8f aq0 = {}, aq1 = {}, ak0 = {}, ak1 = {};
  for (int c0 = 0; c0 < CCH; c0 += 32) {
    v16bf Aq, Ak, Bq0, Bq1, Bk0, Bk1;
#pragma unroll
    for (int h = 0; h < 16; ++h) {
      const int ca = c0 + a_kidx(h, g);                  // A: X^T[i, c]
      Aq[h] = (__bf16)xo[(size_t)ca * NPIX + i0 + m15];
      Ak[h] = (__bf16)xd[(size_t)ca * NPIX + i0 + m15];
      const int cb = c0 + g * 16 + h;                    // B: Wq^T[c, d] = Wq[d, c]
      Bq0[h] = (__bf16)Wq[(size_t)m15 * CCH + cb];
      Bq1[h] = (__bf16)Wq[(size_t)(m15 + 16) * CCH + cb];
      Bk0[h] = (__bf16)Wk[(size_t)m15 * CCH + cb];
      Bk1[h] = (__bf16)Wk[(size_t)(m15 + 16) * CCH + cb];
    }
    aq0 = __builtin_amdgcn_wmma_f32_16x16x32_bf16(false, Aq, false, Bq0, (short)0, aq0, false, false);
    aq1 = __builtin_amdgcn_wmma_f32_16x16x32_bf16(false, Aq, false, Bq1, (short)0, aq1, false, false);
    ak0 = __builtin_amdgcn_wmma_f32_16x16x32_bf16(false, Ak, false, Bk0, (short)0, ak0, false, false);
    ak1 = __builtin_amdgcn_wmma_f32_16x16x32_bf16(false, Ak, false, Bk1, (short)0, ak1, false, false);
  }
  const float bq0v = bq[m15], bq1v = bq[m15 + 16];
  const float bk0v = bk[m15], bk1v = bk[m15 + 16];
#pragma unroll
  for (int r = 0; r < 8; ++r) {
    const int i = i0 + r + 8 * g;
    const size_t base = ((size_t)b * NPIX + i) * DQK;
    qT[base + m15]      = (__bf16)(aq0[r] + bq0v);
    qT[base + 16 + m15] = (__bf16)(aq1[r] + bq1v);
    kT[base + m15]      = (__bf16)(ak0[r] + bk0v);
    kT[base + 16 + m15] = (__bf16)(ak1[r] + bk1v);
  }
}

// ---------------- projection: V = Wv X_dem, channel-major [C,N] ----------------
__global__ void __launch_bounds__(32) proj_v_kernel(
    const float* __restrict__ dem, const float* __restrict__ Wv,
    const float* __restrict__ bv, __bf16* __restrict__ V)
{
  const int lane = threadIdx.x & 31;
  const int m15  = lane & 15;
  const int g    = lane >> 4;
  const int b    = blockIdx.z;
  const int c0   = blockIdx.y * 16;   // output channel tile
  const int j0   = blockIdx.x * 64;   // 4 pixel tiles
  const float* xd = dem + (size_t)b * CCH * NPIX;

  v8f acc[4] = {};
  for (int k0 = 0; k0 < CCH; k0 += 32) {
    v16bf A;
#pragma unroll
    for (int h = 0; h < 16; ++h)
      A[h] = (__bf16)Wv[(size_t)(c0 + m15) * CCH + k0 + a_kidx(h, g)];
#pragma unroll
    for (int t = 0; t < 4; ++t) {
      v16bf Bf;
#pragma unroll
      for (int h = 0; h < 16; ++h)
        Bf[h] = (__bf16)xd[(size_t)(k0 + g * 16 + h) * NPIX + j0 + t * 16 + m15];
      acc[t] = __builtin_amdgcn_wmma_f32_16x16x32_bf16(false, A, false, Bf, (short)0, acc[t], false, false);
    }
  }
#pragma unroll
  for (int r = 0; r < 8; ++r) {
    const int c = c0 + r + 8 * g;
    const float bvv = bv[c];
#pragma unroll
    for (int t = 0; t < 4; ++t)
      V[(size_t)b * CCH * NPIX + (size_t)c * NPIX + j0 + t * 16 + m15] = (__bf16)(acc[t][r] + bvv);
  }
}

// ------- flash attention with TDM double-buffered K/V staging into LDS -------
__global__ void __launch_bounds__(32) attn_kernel(
    const __bf16* __restrict__ qT, const __bf16* __restrict__ kT,
    const __bf16* __restrict__ V, __bf16* __restrict__ OT)
{
  __shared__ alignas(64) __bf16 sK[2][32 * 32];    //  2 x 2 KB : K chunk [j(32)][d(32)]
  __shared__ alignas(64) __bf16 sV[2][CCH * 32];   //  2 x 16 KB: V chunk [c(256)][j(32)]
  __shared__ alignas(64) __bf16 sP[16 * 32];       //  1 KB     : P tile relayout

  const int lane = threadIdx.x & 31;
  const int m15  = lane & 15;
  const int g    = lane >> 4;
  const int b    = blockIdx.y;
  const int i0   = blockIdx.x * 16;
  const __bf16* qTb = qT + (size_t)b * NPIX * DQK;
  const __bf16* kTb = kT + (size_t)b * NPIX * DQK;
  const __bf16* Vb  = V  + (size_t)b * CCH * NPIX;

  // D# group1 for K chunk: 1D row of 1024 bf16 (2 KB contiguous)
  u32x8 g1k;
  g1k[0] = 0x00010000u;                       // data_size = 2B
  g1k[1] = (131072u & 0xFFFFu) << 16;         // tensor_dim0 lo16 (=0)
  g1k[2] = (131072u >> 16) | (1u << 16);      // tensor_dim0 hi16 | tensor_dim1 lo16
  g1k[3] = 0u | (1024u << 16);                // tensor_dim1 hi16 | tile_dim0
  g1k[4] = 1u;                                // tile_dim1 | tile_dim2=0
  g1k[5] = 131072u;                           // tensor_dim0_stride lo32
  g1k[6] = 0u; g1k[7] = 0u;

  // D# group1 for V chunk: 2D tile 256 rows x 32 elems, row stride 4096
  u32x8 g1v;
  g1v[0] = 0x00010000u;                       // data_size = 2B
  g1v[1] = (4096u & 0xFFFFu) << 16;           // tensor_dim0 lo16
  g1v[2] = (4096u >> 16) | (256u << 16);      // tensor_dim0 hi16 | tensor_dim1 lo16
  g1v[3] = 0u | (32u << 16);                  // tensor_dim1 hi16 | tile_dim0
  g1v[4] = 256u;                              // tile_dim1=256 | tile_dim2=0
  g1v[5] = 4096u;                             // tensor_dim0_stride lo32
  g1v[6] = 0u; g1v[7] = 0u;

  v16bf Aq;                                   // A fragment of q^T rows i0..i0+15, K = d
#pragma unroll
  for (int h = 0; h < 16; ++h)
    Aq[h] = qTb[(size_t)(i0 + m15) * DQK + a_kidx(h, g)];

  float m_run[8], l_run[8];
#pragma unroll
  for (int r = 0; r < 8; ++r) { m_run[r] = -1.0e30f; l_run[r] = 0.0f; }
  v8f oacc[16] = {};

  // prologue: stage chunk 0 into buffer 0
  tdm_issue(LDS_OFF(&sK[0][0]), (unsigned long long)(kTb), g1k);
  tdm_issue(LDS_OFF(&sV[0][0]), (unsigned long long)(Vb), g1v);

  for (int j0 = 0; j0 < NPIX; j0 += 32) {
    const int buf = (j0 >> 5) & 1;
    if (j0 + 32 < NPIX) {                     // stage next chunk, then wait for current
      tdm_issue(LDS_OFF(&sK[buf ^ 1][0]),
                (unsigned long long)(kTb + (size_t)(j0 + 32) * DQK), g1k);
      tdm_issue(LDS_OFF(&sV[buf ^ 1][0]),
                (unsigned long long)(Vb + (j0 + 32)), g1v);
      asm volatile("s_wait_tensorcnt 0x2" ::: "memory");  // TDM in-order: current done
    } else {
      asm volatile("s_wait_tensorcnt 0x0" ::: "memory");
    }
    const __bf16* Kb = &sK[buf][0];
    const __bf16* Vs = &sV[buf][0];

    // S tiles: B fragment lane holds column j, K=d contiguous 32B in LDS
    const v16bf Bk0 = *(const v16bf*)(Kb + m15 * 32 + g * 16);
    const v16bf Bk1 = *(const v16bf*)(Kb + (16 + m15) * 32 + g * 16);
    v8f zero = {};
    v8f S0 = __builtin_amdgcn_wmma_f32_16x16x32_bf16(false, Aq, false, Bk0, (short)0, zero, false, false);
    v8f S1 = __builtin_amdgcn_wmma_f32_16x16x32_bf16(false, Aq, false, Bk1, (short)0, zero, false, false);

    float scale[8];
#pragma unroll
    for (int r = 0; r < 8; ++r) {             // row (= r + 8*g) lives across 16 lanes
      float v = fmaxf(S0[r], S1[r]);
#pragma unroll
      for (int off = 1; off < 16; off <<= 1)
        v = fmaxf(v, __shfl_xor(v, off, 32));
      const float mn = fmaxf(m_run[r], v);
      scale[r] = __expf(m_run[r] - mn);
      m_run[r] = mn;
      S0[r] = __expf(S0[r] - mn);
      S1[r] = __expf(S1[r] - mn);
      float s = S0[r] + S1[r];
#pragma unroll
      for (int off = 1; off < 16; off <<= 1)
        s += __shfl_xor(s, off, 32);
      l_run[r] = l_run[r] * scale[r] + s;
    }
#pragma unroll
    for (int t = 0; t < 16; ++t)
#pragma unroll
      for (int r = 0; r < 8; ++r)
        oacc[t][r] *= scale[r];

    // stage P (C/D layout) -> LDS -> reload in A-fragment layout
#pragma unroll
    for (int r = 0; r < 8; ++r) {
      sP[(r + 8 * g) * 32 + m15]      = (__bf16)S0[r];
      sP[(r + 8 * g) * 32 + 16 + m15] = (__bf16)S1[r];
    }
    __syncthreads();
    v16bf Ap;
#pragma unroll
    for (int h = 0; h < 16; ++h)
      Ap[h] = sP[m15 * 32 + a_kidx(h, g)];
    __syncthreads();

#pragma unroll
    for (int t = 0; t < 16; ++t) {            // B fragment of V^T from LDS: contiguous 32B
      const v16bf Bv = *(const v16bf*)(Vs + (t * 16 + m15) * 32 + g * 16);
      oacc[t] = __builtin_amdgcn_wmma_f32_16x16x32_bf16(false, Ap, false, Bv, (short)0, oacc[t], false, false);
    }
  }
#pragma unroll
  for (int r = 0; r < 8; ++r) {
    const float inv = 1.0f / l_run[r];
    const size_t base = ((size_t)b * NPIX + i0 + r + 8 * g) * CCH;
#pragma unroll
    for (int t = 0; t < 16; ++t)
      OT[base + t * 16 + m15] = (__bf16)(oacc[t][r] * inv);
  }
}

// ---------------- gate + skip + mix, f32 output [B,C,H,W] ----------------
__global__ void __launch_bounds__(256) fuse_kernel(
    const float* __restrict__ opt, const float* __restrict__ dem,
    const float* __restrict__ optc, const float* __restrict__ demc,
    const float* __restrict__ gamma, const float* __restrict__ alpha,
    const float* __restrict__ beta, const __bf16* __restrict__ OT,
    float* __restrict__ out)
{
  const size_t total = (size_t)NB * CCH * NPIX;
  const size_t idx = (size_t)blockIdx.x * blockDim.x + threadIdx.x;
  if (idx >= total) return;
  const int i = (int)(idx % NPIX);
  const int b = (int)(idx / ((size_t)CCH * NPIX));
  const float gm = gamma[0], al = alpha[0], be = beta[0];
  const float s = opt[idx] + dem[idx];
  const float x = al * optc[(size_t)b * NPIX + i] + be * demc[(size_t)b * NPIX + i];
  const float G = 1.0f / (1.0f + __expf(-x));
  const int c = (int)((idx / NPIX) % CCH);
  const float o = (float)OT[((size_t)b * NPIX + i) * CCH + c];
  out[idx] = G * (gm * o + (1.0f - gm) * s) + (1.0f - G) * s;
}

extern "C" void kernel_launch(void* const* d_in, const int* in_sizes, int n_in,
                              void* d_out, int out_size, void* d_ws, size_t ws_size,
                              hipStream_t stream)
{
  const float* opt   = (const float*)d_in[0];
  const float* dem   = (const float*)d_in[1];
  const float* optc  = (const float*)d_in[2];
  const float* demc  = (const float*)d_in[3];
  const float* Wq    = (const float*)d_in[4];
  const float* bq    = (const float*)d_in[5];
  const float* Wk    = (const float*)d_in[6];
  const float* bk    = (const float*)d_in[7];
  const float* Wv    = (const float*)d_in[8];
  const float* bv    = (const float*)d_in[9];
  const float* gamma = (const float*)d_in[10];
  const float* alpha = (const float*)d_in[11];
  const float* beta  = (const float*)d_in[12];
  float* out = (float*)d_out;

  char* ws = (char*)d_ws;
  __bf16* qT = (__bf16*)(ws);                                  //  2 MB
  __bf16* kT = (__bf16*)(ws + (size_t)2  * 1024 * 1024);       //  2 MB
  __bf16* V  = (__bf16*)(ws + (size_t)4  * 1024 * 1024);       // 16 MB
  __bf16* OT = (__bf16*)(ws + (size_t)20 * 1024 * 1024);       // 16 MB

  dim3 gqk(NPIX / 16, NB);
  proj_qk_kernel<<<gqk, 32, 0, stream>>>(opt, dem, Wq, bq, Wk, bk, qT, kT);

  dim3 gv(NPIX / 64, CCH / 16, NB);
  proj_v_kernel<<<gv, 32, 0, stream>>>(dem, Wv, bv, V);

  dim3 ga(NPIX / 16, NB);
  attn_kernel<<<ga, 32, 0, stream>>>(qT, kT, V, OT);

  const size_t total = (size_t)NB * CCH * NPIX;
  fuse_kernel<<<(unsigned)((total + 255) / 256), 256, 0, stream>>>(
      opt, dem, optc, demc, gamma, alpha, beta, OT, out);
}
